// LCN_57853209477595
// MI455X (gfx1250) — matrix operand order repaired and checked by reading, not compiled, not run on hardware
//
#include <hip/hip_runtime.h>
#include <hip/hip_bf16.h>

typedef __attribute__((ext_vector_type(16))) _Float16 v16h;
typedef __attribute__((ext_vector_type(8)))  _Float16 v8h;
typedef __attribute__((ext_vector_type(8)))  float    v8f;
typedef __attribute__((ext_vector_type(4)))  unsigned int v4u;
typedef __attribute__((ext_vector_type(8)))  int      v8i;
typedef __attribute__((ext_vector_type(4)))  int      v4i;

#define BATCH   512
#define HH      280
#define WW      280
#define KK      28
#define HSPAN   10
#define WSPAN   10
#define SPAN    100      // HSPAN*WSPAN
#define NF      10       // filters
#define PATCH   784      // 28*28
#define DEC_IN  1000     // NF*SPAN
#define NOUT    10

// ---------------------------------------------------------------------------
// Kernel 1: per-location GEMM  patches[16,784] x w_s[784,16(pad)] -> [16,16]
// One wave32 per (batch-tile, location). The 16x28x28 f32 patch tile is DMA'd
// into LDS by the Tensor Data Mover (3D tile: 28 cols, 28 rows stride 280,
// 16 batches stride 78400), then A fragments are built with aligned
// ds_load_b128 reads. bias + ReLU fused; f16 activation out.
// ---------------------------------------------------------------------------
__global__ __launch_bounds__(32) void lcn_conv_wmma(
    const float* __restrict__ x,       // [B,1,280,280]
    const float* __restrict__ conv_w,  // [F*SPAN,1,28,28]
    const float* __restrict__ conv_b,  // [F*SPAN,1]
    _Float16* __restrict__ y)          // [B,1000] activation (f16)
{
    // LDS: 16 batches x 784 patch-linear f32 (contiguous, as TDM writes it)
    // + 16-float zero tail so chunk-24 reads (k=784..799) of batch 15 are finite.
    __shared__ float smem[16 * PATCH + 16];

    const int lane = threadIdx.x;                   // 32-thread block = 1 wave
    const int tile = blockIdx.x;                    // 0..3199
    const int s  = tile % SPAN;
    const int bt = tile / SPAN;                     // 0..31
    const int b0 = bt * 16;
    const int hs = s / WSPAN, ws = s % WSPAN;

    const int m    = lane & 15;                     // A row / D row-group base
    const int half = lane >> 4;                     // lane-group (K phase)
    const int n    = lane & 15;                     // B/D column this lane owns

    // zero the 16 uninitialized floats past the TDM-written region
    if (lane < 16) smem[16 * PATCH + lane] = 0.0f;

    // ---- Tensor DMA: global 3D tile -> LDS (ISA D# layout, §8.3-8.6) ----
    {
        const unsigned long long gaddr =
            (unsigned long long)(const void*)(x + (size_t)b0 * (HH * WW)
                                              + hs * KK * WW + ws * KK);
        const unsigned lds_addr = (unsigned)(uintptr_t)&smem[0];

        v4u g0;
        g0[0] = 1u;                                  // count=1 valid descriptor
        g0[1] = lds_addr;                            // bits 63:32  lds_addr
        g0[2] = (unsigned)(gaddr & 0xFFFFFFFFu);     // bits 95:64  gaddr lo
        g0[3] = (unsigned)((gaddr >> 32) & 0x1FFFFFFu) | (2u << 30); // type=2

        v8i g1;
        g1[0] = (int)(2u << 16);                     // data_size=2 -> 4 bytes
        g1[1] = (int)(28u << 16);                    // tensor_dim0 = 28
        g1[2] = (int)(28u << 16);                    // tensor_dim1 = 28
        g1[3] = (int)(28u << 16);                    // tile_dim0   = 28
        g1[4] = (int)(28u | (16u << 16));            // tile_dim1=28, tile_dim2=16
        g1[5] = (int)280;                            // tensor_dim0_stride = 280
        g1[6] = (int)((78400u & 0xFFFFu) << 16);     // tensor_dim1_stride lo16
        g1[7] = (int)(78400u >> 16);                 // tensor_dim1_stride hi
        v4i g2;
        g2[0] = 16;                                  // tensor_dim2 = 16
        g2[1] = 0; g2[2] = 0; g2[3] = 0;             // no dim3, no iterate
        v4i g3;
        g3[0] = 0; g3[1] = 0; g3[2] = 0; g3[3] = 0;  // no dim3/dim4
        v8i g4;                                      // extra operand of 6-arg
        g4[0] = 0; g4[1] = 0; g4[2] = 0; g4[3] = 0;  // form (clang-23); unused
        g4[4] = 0; g4[5] = 0; g4[6] = 0; g4[7] = 0;  // fields -> zero

        __builtin_amdgcn_tensor_load_to_lds(g0, g1, g2, g3, g4, 0);
        __builtin_amdgcn_s_wait_tensorcnt(0);        // LDS tile ready
    }

    const float* ap = &smem[m * PATCH];              // this lane's A row
    const float* pw = conv_w + (size_t)(n * SPAN + s) * PATCH;

    v8f c = {};
    for (int kc = 0; kc < 25; ++kc) {                // 25*32 = 800 >= 784
        const int kbase = kc * 32;

        // ---- A fragment from LDS: two contiguous 8-f32 runs per lane ----
        // A 16x32 layout (contiguous, matches B): element e -> k=half*16+e
        v16h a;
#pragma unroll
        for (int h = 0; h < 2; ++h) {
            const float4 q0 = *(const float4*)(ap + kbase + half * 16 + h * 8);
            const float4 q1 = *(const float4*)(ap + kbase + half * 16 + h * 8 + 4);
            a[h * 8 + 0] = (_Float16)q0.x;
            a[h * 8 + 1] = (_Float16)q0.y;
            a[h * 8 + 2] = (_Float16)q0.z;
            a[h * 8 + 3] = (_Float16)q0.w;
            a[h * 8 + 4] = (_Float16)q1.x;
            a[h * 8 + 5] = (_Float16)q1.y;
            a[h * 8 + 6] = (_Float16)q1.z;
            a[h * 8 + 7] = (_Float16)q1.w;
        }

        // ---- B fragment: lane = column n, 16 consecutive k (float4 loads) ----
        // k >= 784 stays zero => LDS tail garbage in A is multiplied by 0.
        v16h b;
        const int kb = kbase + (half << 4);
#pragma unroll
        for (int g = 0; g < 4; ++g) {
            const int k4 = kb + g * 4;               // 784 % 4 == 0 -> aligned
            float4 wv = make_float4(0.f, 0.f, 0.f, 0.f);
            if (n < NF && k4 < PATCH)
                wv = *(const float4*)(pw + k4);
            b[g * 4 + 0] = (_Float16)wv.x;
            b[g * 4 + 1] = (_Float16)wv.y;
            b[g * 4 + 2] = (_Float16)wv.z;
            b[g * 4 + 3] = (_Float16)wv.w;
        }

        c = __builtin_amdgcn_wmma_f32_16x16x32_f16(
                /*neg_a=*/false, a, /*neg_b=*/false, b,
                /*c_mod=*/(short)0, c, /*reuse_a=*/false, /*reuse_b=*/false);
    }

    // ---- epilogue: bias + ReLU, store f16 activation y[b, n*100 + s] ----
    if (n < NF) {
        const float bias = conv_b[n * SPAN + s];
#pragma unroll
        for (int r = 0; r < 8; ++r) {
            const int mm = (half << 3) + r;          // D: M = r + 8*half
            float v = c[r] + bias;
            v = v > 0.0f ? v : 0.0f;
            y[(size_t)(b0 + mm) * DEC_IN + n * SPAN + s] = (_Float16)v;
        }
    }
}

// ---------------------------------------------------------------------------
// Kernel 2: decoder  y[512,1000] x dec_w^T[1000,10] + dec_b -> out[512,10]
// One wave32 per 16-batch tile (32 tiles), K = 1000 padded to 1024.
// A fragment = two contiguous 8xf16 runs per lane -> aligned b128 loads
// (row stride 2000 B and run offsets are 16B-aligned; 1000 % 8 == 0 so a
// run is either fully in-bounds or fully padded with zeros).
// ---------------------------------------------------------------------------
__global__ __launch_bounds__(256) void lcn_dec_wmma(
    const _Float16* __restrict__ y,     // [B,1000] f16
    const float* __restrict__ dec_w,    // [10,1000]
    const float* __restrict__ dec_b,    // [10]
    float* __restrict__ out)            // [B,10]
{
    const int lane = threadIdx.x & 31;
    const int wave = threadIdx.x >> 5;
    const int bt   = blockIdx.x * 8 + wave;          // 0..31 (grid exact)
    const int b0   = bt * 16;

    const int m    = lane & 15;
    const int half = lane >> 4;
    const int n    = lane & 15;

    const _Float16* py = y + (size_t)(b0 + m) * DEC_IN;
    const float*    pw = dec_w + (size_t)n * DEC_IN;

    v8f c = {};
    for (int kc = 0; kc < 32; ++kc) {                // 32*32 = 1024 >= 1000
        const int kbase = kc * 32;

        // ---- A fragment: element e -> k=kbase+half*16+e ----
        v16h a;
#pragma unroll
        for (int h = 0; h < 2; ++h) {
            const int kb8 = kbase + (half << 4) + h * 8;
            v8h q = {};
            if (kb8 < DEC_IN)                        // whole 8-run valid or not
                q = *(const v8h*)(py + kb8);
#pragma unroll
            for (int e = 0; e < 8; ++e)
                a[h * 8 + e] = q[e];
        }

        // ---- B fragment: lane = column n, 16 consecutive k (float4 loads) ----
        v16h b;
        const int kb = kbase + (half << 4);
#pragma unroll
        for (int g = 0; g < 4; ++g) {
            const int k4 = kb + g * 4;               // 1000 % 4 == 0 -> aligned
            float4 wv = make_float4(0.f, 0.f, 0.f, 0.f);
            if (n < NOUT && k4 < DEC_IN)
                wv = *(const float4*)(pw + k4);
            b[g * 4 + 0] = (_Float16)wv.x;
            b[g * 4 + 1] = (_Float16)wv.y;
            b[g * 4 + 2] = (_Float16)wv.z;
            b[g * 4 + 3] = (_Float16)wv.w;
        }

        c = __builtin_amdgcn_wmma_f32_16x16x32_f16(
                false, a, false, b, (short)0, c, false, false);
    }

    if (n < NOUT) {
        const float bias = dec_b[n];
#pragma unroll
        for (int r = 0; r < 8; ++r) {
            const int mm = (half << 3) + r;
            out[(size_t)(b0 + mm) * NOUT + n] = c[r] + bias;
        }
    }
}

// ---------------------------------------------------------------------------
extern "C" void kernel_launch(void* const* d_in, const int* in_sizes, int n_in,
                              void* d_out, int out_size, void* d_ws, size_t ws_size,
                              hipStream_t stream) {
    const float* x      = (const float*)d_in[0];   // 512*1*280*280
    const float* conv_w = (const float*)d_in[1];   // 1000*1*28*28
    const float* conv_b = (const float*)d_in[2];   // 1000*1
    const float* dec_w  = (const float*)d_in[3];   // 10*1000
    const float* dec_b  = (const float*)d_in[4];   // 10
    float* outp = (float*)d_out;                   // 512*10

    _Float16* y = (_Float16*)d_ws;                 // 512*1000 f16 = 1,024,000 B

    // 3200 tiles (32 batch-tiles x 100 locations), one wave32 + TDM each
    lcn_conv_wmma<<<3200, 32, 0, stream>>>(x, conv_w, conv_b, y);
    // 32 batch-tiles, 8 waves per block
    lcn_dec_wmma<<<4, 256, 0, stream>>>(y, dec_w, dec_b, outp);
}
